// EEGGNN_73967926772095
// MI455X (gfx1250) — matrix operand order repaired and checked by reading, not compiled, not run on hardware
//
#include <hip/hip_runtime.h>

typedef __attribute__((ext_vector_type(16))) _Float16 v16h;
typedef __attribute__((ext_vector_type(8)))  _Float16 v8h;
typedef __attribute__((ext_vector_type(8)))  float    v8f;

#define N_NODES   100000
#define N_EDGES   640000
#define FEAT      128
#define HIDDEN    64
#define NUM_GRAPHS 256

// ---------------- kernel 1: h = x (init accumulator) ----------------
__global__ __launch_bounds__(256) void k_init_h(const float* __restrict__ x,
                                                float* __restrict__ h, int n4) {
  int i = blockIdx.x * blockDim.x + threadIdx.x;
  if (i < n4) ((float4*)h)[i] = ((const float4*)x)[i];
}

// ---------------- kernel 2: zero pooled sums / counts ----------------
__global__ __launch_bounds__(256) void k_zero_pool(float* __restrict__ sums,
                                                   float* __restrict__ counts) {
  int i = blockIdx.x * blockDim.x + threadIdx.x;
  if (i < NUM_GRAPHS * HIDDEN) sums[i] = 0.0f;
  if (i < NUM_GRAPHS) counts[i] = 0.0f;
}

// ---------------- kernel 3: edge scatter-add (GIN aggregation) ----------------
// one thread per (edge, 4 features): wave of 32 lanes covers one full 128-f row
__global__ __launch_bounds__(256) void k_edge_scatter(const float* __restrict__ x,
                                                      const int* __restrict__ ei,
                                                      float* __restrict__ h) {
  long long t = (long long)blockIdx.x * blockDim.x + threadIdx.x;
  int e = (int)(t >> 5);
  if (e >= N_EDGES) return;
  int c = ((int)t & 31) * 4;
  int src = ei[e];
  int dst = ei[N_EDGES + e];
  float4 v = *(const float4*)(x + (long long)src * FEAT + c);
  float* d = h + (long long)dst * FEAT + c;
  atomicAdd(d + 0, v.x);
  atomicAdd(d + 1, v.y);
  atomicAdd(d + 2, v.z);
  atomicAdd(d + 3, v.w);
}

// ---------------- kernel 4: per-graph node counts ----------------
__global__ __launch_bounds__(256) void k_counts(const int* __restrict__ batch,
                                                float* __restrict__ counts) {
  int i = blockIdx.x * blockDim.x + threadIdx.x;
  if (i < N_NODES) atomicAdd(&counts[batch[i]], 1.0f);
}

// ---------------- kernel 5: fused GIN MLP (WMMA) + pooled accumulation ----------------
// 256 threads = 8 waves; each wave computes one 16-node tile through both GEMMs.
__global__ __launch_bounds__(256) void k_mlp_wmma(const float* __restrict__ h,
                                                  const int* __restrict__ batch,
                                                  const float* __restrict__ W1,
                                                  const float* __restrict__ b1,
                                                  const float* __restrict__ W2,
                                                  const float* __restrict__ b2,
                                                  float* __restrict__ sums) {
  // W1 fragments: 4 ntiles x 4 kchunks x 32 lanes x 16 halves (fragment-ready layout)
  __shared__ __align__(32) _Float16 sW1[4 * 4 * 32 * 16];  // 16 KB
  // W2 fragments: 4 ntiles x 2 kchunks x 32 lanes x 16 halves
  __shared__ __align__(32) _Float16 sW2[4 * 2 * 32 * 16];  //  8 KB
  __shared__ float sb1[HIDDEN];
  __shared__ float sb2[HIDDEN];
  // per-wave 16x64 f16 intermediate (C-layout -> A-layout cross-lane reshuffle)
  __shared__ __align__(32) _Float16 sMid[8][16 * 64];      // 16 KB

  const int tid = threadIdx.x;

  // Stage W1 into B-fragment layout: lane l -> col n=l&15, half j -> K=(j>>3)*16+(l>>4)*8+(j&7)
  for (int idx = tid; idx < 4 * 4 * 32 * 16; idx += 256) {
    int j = idx & 15, lane = (idx >> 4) & 31, kc = (idx >> 9) & 3, nt = idx >> 11;
    int n = lane & 15, hi = lane >> 4;
    int K = kc * 32 + (j >> 3) * 16 + hi * 8 + (j & 7);
    sW1[idx] = (_Float16)W1[K * HIDDEN + nt * 16 + n];
  }
  for (int idx = tid; idx < 4 * 2 * 32 * 16; idx += 256) {
    int j = idx & 15, lane = (idx >> 4) & 31, kc = (idx >> 9) & 1, nt = idx >> 10;
    int n = lane & 15, hi = lane >> 4;
    int K = kc * 32 + (j >> 3) * 16 + hi * 8 + (j & 7);
    sW2[idx] = (_Float16)W2[K * HIDDEN + nt * 16 + n];
  }
  if (tid < HIDDEN) { sb1[tid] = b1[tid]; sb2[tid] = b2[tid]; }
  __syncthreads();

  const int wave = tid >> 5;
  const int lane = tid & 31;
  const int m  = lane & 15;   // row (A) / col (B,C) within 16
  const int hi = lane >> 4;
  const int tile = blockIdx.x * 8 + wave;

  // clamp row index for tail tiles so EXEC stays all-ones through the WMMAs
  int row = tile * 16 + m;
  int rowc = row < N_NODES ? row : (N_NODES - 1);
  const float4* hp4 = (const float4*)(h + (long long)rowc * FEAT);

  // A fragments (16x32 f16): lane m, halves 0..7 -> K = kc*32 + hi*8 + j,
  //                          halves 8..15 -> K = kc*32 + 16 + hi*8 + (j-8)
  v16h A[4];
#pragma unroll
  for (int kc = 0; kc < 4; ++kc) {
    float4 p0 = hp4[kc * 8 + hi * 2];
    float4 p1 = hp4[kc * 8 + hi * 2 + 1];
    float4 p2 = hp4[kc * 8 + 4 + hi * 2];
    float4 p3 = hp4[kc * 8 + 4 + hi * 2 + 1];
    v16h a;
    a[0]  = (_Float16)p0.x; a[1]  = (_Float16)p0.y; a[2]  = (_Float16)p0.z; a[3]  = (_Float16)p0.w;
    a[4]  = (_Float16)p1.x; a[5]  = (_Float16)p1.y; a[6]  = (_Float16)p1.z; a[7]  = (_Float16)p1.w;
    a[8]  = (_Float16)p2.x; a[9]  = (_Float16)p2.y; a[10] = (_Float16)p2.z; a[11] = (_Float16)p2.w;
    a[12] = (_Float16)p3.x; a[13] = (_Float16)p3.y; a[14] = (_Float16)p3.z; a[15] = (_Float16)p3.w;
    A[kc] = a;
  }

  _Float16* mid = sMid[wave];

  // GEMM1: [16x128] @ [128x64]  -> bias + ReLU -> sMid (f16)
#pragma unroll
  for (int nt = 0; nt < 4; ++nt) {
    v8f acc = {};
#pragma unroll
    for (int kc = 0; kc < 4; ++kc) {
      v16h bfr = *(const v16h*)&sW1[((nt * 4 + kc) * 32 + lane) * 16];
      acc = __builtin_amdgcn_wmma_f32_16x16x32_f16(false, A[kc], false, bfr,
                                                   (short)0, acc, false, false);
    }
    float bias = sb1[nt * 16 + m];  // C col = lane&15
#pragma unroll
    for (int r = 0; r < 8; ++r) {
      float v = acc[r] + bias;
      v = v > 0.0f ? v : 0.0f;
      mid[(r + 8 * hi) * HIDDEN + nt * 16 + m] = (_Float16)v;  // C row = r + 8*hi
    }
  }
  __syncthreads();

  // Rebuild A fragments for GEMM2 from sMid (16x64 f16)
  v16h A2[2];
#pragma unroll
  for (int kc = 0; kc < 2; ++kc) {
    v8h q0 = *(const v8h*)&mid[m * HIDDEN + kc * 32 + hi * 8];
    v8h q1 = *(const v8h*)&mid[m * HIDDEN + kc * 32 + 16 + hi * 8];
    v16h a;
#pragma unroll
    for (int j = 0; j < 8; ++j) { a[j] = q0[j]; a[j + 8] = q1[j]; }
    A2[kc] = a;
  }

  // per-output-row graph ids (predicate only the atomic stores, not the WMMAs)
  int garr[8];
  bool vld[8];
#pragma unroll
  for (int r = 0; r < 8; ++r) {
    int node = tile * 16 + r + 8 * hi;
    vld[r] = node < N_NODES;
    garr[r] = vld[r] ? batch[node] : 0;
  }

  // GEMM2: [16x64] @ [64x64] -> bias + ReLU -> atomic pooled accumulation
#pragma unroll
  for (int nt = 0; nt < 4; ++nt) {
    v8f acc = {};
#pragma unroll
    for (int kc = 0; kc < 2; ++kc) {
      v16h bfr = *(const v16h*)&sW2[((nt * 2 + kc) * 32 + lane) * 16];
      acc = __builtin_amdgcn_wmma_f32_16x16x32_f16(false, A2[kc], false, bfr,
                                                   (short)0, acc, false, false);
    }
    float bias = sb2[nt * 16 + m];
#pragma unroll
    for (int r = 0; r < 8; ++r) {
      float v = acc[r] + bias;
      v = v > 0.0f ? v : 0.0f;
      if (vld[r]) atomicAdd(&sums[garr[r] * HIDDEN + nt * 16 + m], v);
    }
  }
}

// ---------------- kernel 6: mean-pool divide + classifier head ----------------
__global__ __launch_bounds__(256) void k_head(const float* __restrict__ sums,
                                              const float* __restrict__ counts,
                                              const float* __restrict__ Wc,
                                              const float* __restrict__ bc,
                                              float* __restrict__ out) {
  int t = blockIdx.x * blockDim.x + threadIdx.x;
  if (t >= NUM_GRAPHS * 2) return;
  int g = t >> 1, o = t & 1;
  float cnt = counts[g];
  cnt = cnt > 1.0f ? cnt : 1.0f;
  float inv = 1.0f / cnt;
  float acc = bc[o];
#pragma unroll 8
  for (int k = 0; k < HIDDEN; ++k)
    acc += sums[g * HIDDEN + k] * inv * Wc[k * 2 + o];
  out[t] = acc;
}

extern "C" void kernel_launch(void* const* d_in, const int* in_sizes, int n_in,
                              void* d_out, int out_size, void* d_ws, size_t ws_size,
                              hipStream_t stream) {
  const float* x    = (const float*)d_in[0];
  const int*   ei   = (const int*)d_in[1];
  const int*   batch= (const int*)d_in[2];
  const float* W1   = (const float*)d_in[3];
  const float* b1   = (const float*)d_in[4];
  const float* W2   = (const float*)d_in[5];
  const float* b2   = (const float*)d_in[6];
  const float* Wc   = (const float*)d_in[7];
  const float* bc   = (const float*)d_in[8];
  float* out = (float*)d_out;

  // workspace layout: h accumulator (51.2 MB) | pooled sums (64 KB) | counts (1 KB)
  float* h      = (float*)d_ws;
  float* sums   = h + (size_t)N_NODES * FEAT;
  float* counts = sums + NUM_GRAPHS * HIDDEN;

  int n4 = N_NODES * FEAT / 4;
  k_init_h<<<(n4 + 255) / 256, 256, 0, stream>>>(x, h, n4);
  k_zero_pool<<<(NUM_GRAPHS * HIDDEN + 255) / 256, 256, 0, stream>>>(sums, counts);

  long long et = (long long)N_EDGES * 32;
  k_edge_scatter<<<(int)((et + 255) / 256), 256, 0, stream>>>(x, ei, h);
  k_counts<<<(N_NODES + 255) / 256, 256, 0, stream>>>(batch, counts);

  k_mlp_wmma<<<(N_NODES + 127) / 128, 256, 0, stream>>>(h, batch, W1, b1, W2, b2, sums);
  k_head<<<2, 256, 0, stream>>>(sums, counts, Wc, bc, out);
}
